// EdgeModel_44693429682900
// MI455X (gfx1250) — matrix operand order repaired and checked by reading, not compiled, not run on hardware
//
#include <hip/hip_runtime.h>
#include <hip/hip_bf16.h>

// ---------------------------------------------------------------------------
// CDNA5 (gfx1250) fused edge-MLP:  x = [node[src] | node[dst] | edge | glob]
//   h = ssp(x W1 + b1); h = ssp(h W2 + b2); out = ssp(h W3 + b3)
// bf16 WMMA (v_wmma_f32_16x16x32_bf16) with f32 accumulate, wave32.
// Software-pipelined: B fragments (L2-resident packed weights) and A
// fragments (LDS) are double-buffered one K-slice ahead of the WMMAs.
// ---------------------------------------------------------------------------

typedef __attribute__((ext_vector_type(16))) __bf16 v16bf;
typedef __attribute__((ext_vector_type(8)))  float  v8f;

union Frag { uint4 u[2]; v16bf v; };

__device__ __forceinline__ unsigned short f2bf(float f) {
    unsigned u = __float_as_uint(f);
    u += 0x7FFFu + ((u >> 16) & 1u);          // round-to-nearest-even
    return (unsigned short)(u >> 16);
}

// packed f32x2 -> bf16x2 (v_cvt_pk_bf16_f32)
__device__ __forceinline__ unsigned pkbf2(float x, float y) {
    __hip_bfloat162 t = __float22bfloat162_rn(make_float2(x, y));
    return *reinterpret_cast<unsigned*>(&t);
}

__device__ __forceinline__ float ssp(float x) { // softplus(x) - ln2, stable
    float e = __expf(-fabsf(x));
    return fmaxf(x, 0.0f) + log1pf(e) - 0.69314718056f;
}

// ---------------------------------------------------------------------------
// Weight packer: W [K x nOut] fp32 row-major  ->  bf16 B-fragments.
// Fragment f = nt*numKS + ks is a 32(K) x 16(N) tile; per-lane 16 bf16
// contiguous (32 B) so the main kernel loads it with two b128 loads.
// B layout (16-bit, 32x16): lanes 0-15 hold K=0..7,16..23 (2/VGPR), lanes 16-31
// hold K=8..15,24..31, N = lane % 16.
// ---------------------------------------------------------------------------
__global__ void pack_weights(const float* __restrict__ W,
                             unsigned short* __restrict__ ws,
                             int numKS, int nOut) {
    int t = blockIdx.x * blockDim.x + threadIdx.x;
    int numNT = nOut >> 4;
    if (t >= numKS * numNT * 32) return;
    int f  = t >> 5;
    int l  = t & 31;
    int ks = f % numKS;
    int nt = f / numKS;
    int n     = nt * 16 + (l & 15);
    int kbase = ks * 32 + 8 * (l >> 4);
    unsigned short* d = ws + f * 512 + l * 16;   // 512 bf16 per fragment
    #pragma unroll
    for (int j = 0; j < 16; ++j)
        d[j] = f2bf(W[(long long)(kbase + j + (j & 8)) * nOut + n]);
}

// ---------------------------------------------------------------------------
// One MLP layer for one wave's 16-row M-tile, 4 N-tiles per group.
// A (16-bit, 16x32): lane<16: K runs [ks*32+0..7] & [ks*32+16..23];
//                    lane>=16: runs shifted by +8.  Two ds_load_b128 per frag.
// ---------------------------------------------------------------------------
template<int NUM_KS, int NUM_NT, bool TO_GLOBAL>
__device__ __forceinline__ void mlp_layer(
    const unsigned short* __restrict__ sSrc, int srcStride,
    const unsigned short* __restrict__ wsB,
    const float* __restrict__ bias,
    unsigned short* __restrict__ sDst, int dstStride,
    float* __restrict__ gOut, int gRow0, int gRowMax, int gCols,
    int lane, int row0)
{
    const int sub = lane >> 4;          // half-wave id
    const int col = lane & 15;
    const unsigned short* aRow  = sSrc + (row0 + col) * srcStride;
    const unsigned short* wLane = wsB + lane * 16;

    for (int ng = 0; ng < NUM_NT / 4; ++ng) {
        v8f acc[4];
        #pragma unroll
        for (int j = 0; j < 4; ++j) {
            float b = bias[(ng * 4 + j) * 16 + col];
            #pragma unroll
            for (int v = 0; v < 8; ++v) acc[j][v] = b;
        }

        // prologue: fragments for ks = 0
        Frag a, b[4];
        a.u[0] = *(const uint4*)(aRow + sub * 8);
        a.u[1] = *(const uint4*)(aRow + 16 + sub * 8);
        #pragma unroll
        for (int j = 0; j < 4; ++j) {
            const uint4* q =
                (const uint4*)(wLane + (size_t)((ng * 4 + j) * NUM_KS) * 512);
            b[j].u[0] = q[0];
            b[j].u[1] = q[1];
        }

        #pragma unroll
        for (int ks = 0; ks < NUM_KS; ++ks) {
            // issue next slice's loads BEFORE this slice's WMMAs
            Frag an, bn[4];
            if (ks + 1 < NUM_KS) {
                an.u[0] = *(const uint4*)(aRow + (ks + 1) * 32 + sub * 8);
                an.u[1] = *(const uint4*)(aRow + (ks + 1) * 32 + 16 + sub * 8);
                #pragma unroll
                for (int j = 0; j < 4; ++j) {
                    const uint4* q = (const uint4*)
                        (wLane + (size_t)((ng * 4 + j) * NUM_KS + ks + 1) * 512);
                    bn[j].u[0] = q[0];
                    bn[j].u[1] = q[1];
                }
            }
            #pragma unroll
            for (int j = 0; j < 4; ++j)
                acc[j] = __builtin_amdgcn_wmma_f32_16x16x32_bf16(
                    false, a.v, false, b[j].v, (short)0, acc[j], false, false);
            if (ks + 1 < NUM_KS) {
                a = an;
                #pragma unroll
                for (int j = 0; j < 4; ++j) b[j] = bn[j];
            }
        }

        // activation + store.  C/D layout: VGPR v -> row v + 8*sub, col = lane%16.
        #pragma unroll
        for (int j = 0; j < 4; ++j) {
            const int n = (ng * 4 + j) * 16 + col;
            #pragma unroll
            for (int v = 0; v < 8; ++v) {
                float y = ssp(acc[j][v]);
                int r = row0 + v + 8 * sub;
                if (TO_GLOBAL) {
                    if (gRow0 + r < gRowMax)
                        gOut[(long long)(gRow0 + r) * gCols + n] = y;
                } else {
                    sDst[r * dstStride + n] = f2bf(y);
                }
            }
        }
    }
}

// ---------------------------------------------------------------------------
// Main fused kernel: 128 threads (4 wave32), 64 edges per block.
// LDS: X tile [64][456] bf16 (padded) + H tile [64][264] bf16 + indices.
// ---------------------------------------------------------------------------
#define X_STRIDE 456
#define H_STRIDE 264

__global__ __launch_bounds__(128) void edge_mlp_main(
    const float* __restrict__ node,    // [N,128]
    const float* __restrict__ edgef,   // [E,128]
    const float* __restrict__ globf,   // [G,64]
    const int*   __restrict__ eidx,    // [2,E]
    const int*   __restrict__ batch,   // [N]
    const float* __restrict__ b1, const float* __restrict__ b2,
    const float* __restrict__ b3,
    const unsigned short* __restrict__ wsW1,
    const unsigned short* __restrict__ wsW2,
    const unsigned short* __restrict__ wsW3,
    float* __restrict__ out, int E)
{
    extern __shared__ char smem[];
    unsigned short* sX = (unsigned short*)smem;                        // 64*456
    unsigned short* sH = (unsigned short*)(smem + 64 * X_STRIDE * 2);  // 64*264
    int* sIdx = (int*)(smem + 64 * X_STRIDE * 2 + 64 * H_STRIDE * 2);  // 3*64

    const int tid   = threadIdx.x;
    const int lane  = tid & 31;
    const int wave  = tid >> 5;
    const int ebase = blockIdx.x * 64;

    // stage gather indices
    if (tid < 64) {
        int e = ebase + tid;  if (e >= E) e = E - 1;
        int s = eidx[e];
        sIdx[tid]       = s;
        sIdx[64 + tid]  = eidx[E + e];
        sIdx[128 + tid] = batch[s];
    }
    // warm L2 for the packed weights while indices land
    __builtin_prefetch(wsW1 + (size_t)tid * 64, 0, 1);
    __syncthreads();

    const float4* node4 = (const float4*)node;
    const float4* edge4 = (const float4*)edgef;
    const float4* glob4 = (const float4*)globf;

    // gather + fp32->bf16 convert into the X tile
    for (int i = tid; i < 64 * 32; i += 128) {           // node[src] -> [0,128)
        int r = i >> 5, c = i & 31;
        float4 v = node4[(long long)sIdx[r] * 32 + c];
        *(uint2*)(sX + r * X_STRIDE + c * 4) =
            make_uint2(pkbf2(v.x, v.y), pkbf2(v.z, v.w));
    }
    for (int i = tid; i < 64 * 32; i += 128) {           // node[dst] -> [128,256)
        int r = i >> 5, c = i & 31;
        float4 v = node4[(long long)sIdx[64 + r] * 32 + c];
        *(uint2*)(sX + r * X_STRIDE + 128 + c * 4) =
            make_uint2(pkbf2(v.x, v.y), pkbf2(v.z, v.w));
    }
    for (int i = tid; i < 64 * 32; i += 128) {           // edge -> [256,384)
        int r = i >> 5, c = i & 31;
        int e = ebase + r;  if (e >= E) e = E - 1;
        float4 v = edge4[(long long)e * 32 + c];
        *(uint2*)(sX + r * X_STRIDE + 256 + c * 4) =
            make_uint2(pkbf2(v.x, v.y), pkbf2(v.z, v.w));
    }
    for (int i = tid; i < 64 * 16; i += 128) {           // global -> [384,448)
        int r = i >> 4, c = i & 15;
        float4 v = glob4[(long long)sIdx[128 + r] * 16 + c];
        *(uint2*)(sX + r * X_STRIDE + 384 + c * 4) =
            make_uint2(pkbf2(v.x, v.y), pkbf2(v.z, v.w));
    }
    __syncthreads();

    const int row0 = wave * 16;

    // layer 1: X[64x448] x W1[448x256] -> sH   (14 K-slices, 16 N-tiles)
    mlp_layer<14, 16, false>(sX, X_STRIDE, wsW1, b1,
                             sH, H_STRIDE, nullptr, 0, 0, 0, lane, row0);
    __syncthreads();

    // layer 2: sH[64x256] x W2[256x256] -> sX (reused as h2, stride 264)
    mlp_layer<8, 16, false>(sH, H_STRIDE, wsW2, b2,
                            sX, H_STRIDE, nullptr, 0, 0, 0, lane, row0);
    __syncthreads();

    // layer 3: h2[64x256] x W3[256x128] -> out (fp32, global)
    mlp_layer<8, 8, true>(sX, H_STRIDE, wsW3, b3,
                          nullptr, 0, out, ebase, E, 128, lane, row0);
}

// ---------------------------------------------------------------------------
extern "C" void kernel_launch(void* const* d_in, const int* in_sizes, int n_in,
                              void* d_out, int out_size, void* d_ws, size_t ws_size,
                              hipStream_t stream) {
    const float* node  = (const float*)d_in[0];
    const float* edgef = (const float*)d_in[1];
    const float* globf = (const float*)d_in[2];
    const int*   eidx  = (const int*)d_in[3];
    const int*   batch = (const int*)d_in[4];
    const float* W1 = (const float*)d_in[5];
    const float* b1 = (const float*)d_in[6];
    const float* W2 = (const float*)d_in[7];
    const float* b2 = (const float*)d_in[8];
    const float* W3 = (const float*)d_in[9];
    const float* b3 = (const float*)d_in[10];
    float* out = (float*)d_out;

    const int E = in_sizes[1] / 128;   // edge_feats is [E,128]

    unsigned short* ws   = (unsigned short*)d_ws;
    unsigned short* wsW1 = ws;                         // 14*16 fragments
    unsigned short* wsW2 = wsW1 + 14 * 16 * 512;       // 8*16 fragments
    unsigned short* wsW3 = wsW2 + 8 * 16 * 512;        // 8*8  fragments

    { int t = 14 * 16 * 32; pack_weights<<<(t + 255) / 256, 256, 0, stream>>>(W1, wsW1, 14, 256); }
    { int t = 8 * 16 * 32;  pack_weights<<<(t + 255) / 256, 256, 0, stream>>>(W2, wsW2, 8, 256); }
    { int t = 8 * 8 * 32;   pack_weights<<<(t + 255) / 256, 256, 0, stream>>>(W3, wsW3, 8, 128); }

    const size_t shmem = (size_t)64 * X_STRIDE * 2 + 64 * H_STRIDE * 2 + 3 * 64 * 4;
    const int blocks = (E + 63) / 64;
    edge_mlp_main<<<blocks, 128, shmem, stream>>>(node, edgef, globf, eidx, batch,
                                                  b1, b2, b3, wsW1, wsW2, wsW3,
                                                  out, E);
}
